// OurModel_9835475108494
// MI455X (gfx1250) — matrix operand order repaired and checked by reading, not compiled, not run on hardware
//
#include <hip/hip_runtime.h>
#include <hip/hip_bf16.h>

typedef __attribute__((ext_vector_type(16))) _Float16 v16h;
typedef __attribute__((ext_vector_type(8)))  float    v8f;

#define WMMA_F16(a,b,c) __builtin_amdgcn_wmma_f32_16x16x32_f16(false,(a),false,(b),(short)0,(c),false,false)

// Async global->LDS copy (CDNA5 Tensor/Async path, ASYNCcnt tracked).
// vdst = VGPR with LDS byte offset, vaddr = 64-bit global address, GV mode.
#define ASYNC_B128(ldsoff, gaddr) \
    asm volatile("global_load_async_to_lds_b128 %0, %1, off" :: "v"(ldsoff), "v"(gaddr) : "memory")
#define WAIT_ASYNC(n) asm volatile("s_wait_asynccnt %0" :: "n"(n) : "memory")
#define WAIT_DS0()    asm volatile("s_wait_dscnt 0x0" ::: "memory")

__device__ __forceinline__ unsigned lds_addr(const void* p) {
    // LDS flat addresses: low 32 bits are the LDS byte offset
    return (unsigned)(unsigned long long)p;
}

// ---------------------------------------------------------------------------
// WMMA GEMM with double-buffered async Global->LDS staging.
// C[M,N] = A_h[M,K] * Wh[N,K]^T + bias. A,Wh f16 row-major, 16B-aligned rows
// (K % 32 == 0 -> row stride multiple of 16B). One wave32 per 16x16 tile.
// ---------------------------------------------------------------------------
__global__ __launch_bounds__(32) void gemm_wmma(
    const _Float16* __restrict__ A, const _Float16* __restrict__ Wh,
    const float* __restrict__ bias, float* __restrict__ Cf,
    _Float16* __restrict__ Ch, int M, int N, int K, int relu)
{
    __shared__ __align__(16) _Float16 As[2][16 * 32];
    __shared__ __align__(16) _Float16 Bs[2][16 * 32];   // [n][k] row-major
    const int lane = threadIdx.x;
    const int n0 = blockIdx.x * 16;
    const int m0 = blockIdx.y * 16;
    const int g = lane >> 4, loc = lane & 15;

    // per-lane: 2 x 16B chunks of A tile + 2 x 16B chunks of B tile
    auto issue = [&](int buf, int ke) {
        WAIT_DS0();   // WAR: wave ds_loads of this buffer must have completed
        #pragma unroll
        for (int j = 0; j < 2; ++j) {
            int ch = lane * 2 + j;
            int r = ch >> 2, q = ch & 3;            // row 0..15, 16B chunk 0..3
            unsigned la = lds_addr(&As[buf][0]) + (unsigned)(r * 64 + q * 16);
            unsigned long long ga =
                (unsigned long long)(const void*)(A + (size_t)(m0 + r) * K + ke + q * 8);
            ASYNC_B128(la, ga);
            unsigned lb = lds_addr(&Bs[buf][0]) + (unsigned)(r * 64 + q * 16);
            unsigned long long gb =
                (unsigned long long)(const void*)(Wh + (size_t)(n0 + r) * K + ke + q * 8);
            ASYNC_B128(lb, gb);
        }
    };

    const int nkt = K / 32;
    issue(0, 0);
    v8f c = {};
    for (int kt = 0; kt < nkt; ++kt) {
        int buf = kt & 1;
        if (kt + 1 < nkt) { issue(buf ^ 1, (kt + 1) * 32); WAIT_ASYNC(4); }
        else              { WAIT_ASYNC(0); }
        v16h a, b;
        #pragma unroll
        for (int i = 0; i < 16; ++i) {
            int v = i >> 1, p = i & 1;
            int ka = g * 16 + v * 2 + p;
            a[i] = As[buf][loc * 32 + ka];
            int kb = g * 16 + v * 2 + p;
            b[i] = Bs[buf][loc * 32 + kb];
        }
        c = WMMA_F16(a, b, c);
    }
    float bb = bias ? bias[n0 + loc] : 0.f;
    #pragma unroll
    for (int r = 0; r < 8; ++r) {
        int m = m0 + g * 8 + r;
        float val = c[r] + bb;
        if (relu) val = fmaxf(val, 0.f);
        if (Cf) Cf[(size_t)m * N + n0 + loc] = val;
        if (Ch) Ch[(size_t)m * N + n0 + loc] = (_Float16)val;
    }
}

// ---------------------------------------------------------------------------
// Flash attention over the batch axis (S=512 tokens, per position n, 2 heads,
// hd=48). One wave32 per (query tile, n, h). K/V tiles streamed with
// double-buffered async Global->LDS copies; Q.K^T and P.V via WMMA.
// ---------------------------------------------------------------------------
__global__ __launch_bounds__(32) void attn_wmma(
    const _Float16* __restrict__ qkv, _Float16* __restrict__ oh,
    int S, int L, int E3)
{
    const int hd = 48;
    __shared__ __align__(16) _Float16 Qs[16 * 64];
    __shared__ __align__(16) _Float16 Ks[2][16 * 64];
    __shared__ __align__(16) _Float16 Vs[2][16 * 48];
    __shared__ __align__(16) _Float16 Ps[16 * 16];
    const int lane = threadIdx.x;
    const int qt = blockIdx.x, n = blockIdx.y, h = blockIdx.z;
    const int g = lane >> 4, loc = lane & 15;

    // zero pad region d=48..63 of both K buffers (DMA writes only d<48)
    {
        int r = lane >> 1, q = lane & 1;
        #pragma unroll
        for (int z = 0; z < 8; ++z) {
            Ks[0][r * 64 + 48 + q * 8 + z] = (_Float16)0.f;
            Ks[1][r * 64 + 48 + q * 8 + z] = (_Float16)0.f;
        }
    }
    // stage Q tile (16x64, d>=48 zero)
    #pragma unroll
    for (int j = 0; j < 32; ++j) {
        int e = lane * 32 + j;
        int rq = e >> 6, d = e & 63;
        int s = qt * 16 + rq;
        Qs[e] = (d < hd) ? qkv[(size_t)(s * L + n) * E3 + h * hd + d] : (_Float16)0.f;
    }
    __syncthreads();
    v16h a0, a1;
    #pragma unroll
    for (int i = 0; i < 16; ++i) {
        int v = i >> 1, p = i & 1;
        int ka = g * 16 + v * 2 + p;
        a0[i] = Qs[loc * 64 + ka];
        a1[i] = Qs[loc * 64 + 32 + ka];
    }

    // async stage of one K/V tile: K rows are 96B (6x16B), V rows 96B compact
    auto issueKV = [&](int buf, int kt) {
        WAIT_DS0();
        #pragma unroll
        for (int j = 0; j < 3; ++j) {
            int ch = j * 32 + lane;                 // 0..95
            int r = ch / 6, q = ch % 6;
            int s = kt * 16 + r;
            size_t row = (size_t)(s * L + n) * E3;
            unsigned lk = lds_addr(&Ks[buf][0]) + (unsigned)(r * 128 + q * 16);
            unsigned long long gk =
                (unsigned long long)(const void*)(qkv + row + 96 + h * hd + q * 8);
            ASYNC_B128(lk, gk);
            unsigned lv = lds_addr(&Vs[buf][0]) + (unsigned)(r * 96 + q * 16);
            unsigned long long gv =
                (unsigned long long)(const void*)(qkv + row + 192 + h * hd + q * 8);
            ASYNC_B128(lv, gv);
        }
    };

    float mrow[8], lrow[8];
    v8f o0 = {}, o1 = {}, o2 = {};
    #pragma unroll
    for (int r = 0; r < 8; ++r) { mrow[r] = -1e30f; lrow[r] = 0.f; }
    const float scale = 0.14433756729740646f;   // 1/sqrt(48)

    const int nk = S / 16;
    issueKV(0, 0);
    for (int kt = 0; kt < nk; ++kt) {
        int buf = kt & 1;
        if (kt + 1 < nk) { issueKV(buf ^ 1, kt + 1); WAIT_ASYNC(6); }
        else             { WAIT_ASYNC(0); }
        v16h b0, b1;
        #pragma unroll
        for (int i = 0; i < 16; ++i) {
            int v = i >> 1, p = i & 1;
            int kb = g * 16 + v * 2 + p;             // d index
            b0[i] = Ks[buf][loc * 64 + kb];          // B[k=d][n=key=loc]
            b1[i] = Ks[buf][loc * 64 + 32 + kb];
        }
        v8f sc = {};
        sc = WMMA_F16(a0, b0, sc);
        sc = WMMA_F16(a1, b1, sc);

        float pv[8];
        #pragma unroll
        for (int r = 0; r < 8; ++r) {
            float xv = sc[r] * scale;
            float rm = xv;
            #pragma unroll
            for (int off = 1; off < 16; off <<= 1)
                rm = fmaxf(rm, __shfl_xor(rm, off, 32));
            float mn = fmaxf(mrow[r], rm);
            float alpha = __expf(mrow[r] - mn);
            float p = __expf(xv - mn);
            float rs = p;
            #pragma unroll
            for (int off = 1; off < 16; off <<= 1)
                rs += __shfl_xor(rs, off, 32);
            lrow[r] = lrow[r] * alpha + rs;
            mrow[r] = mn;
            o0[r] *= alpha; o1[r] *= alpha; o2[r] *= alpha;
            pv[r] = p;
        }
        __syncthreads();       // Ps WAR (single-wave: waits only)
        #pragma unroll
        for (int r = 0; r < 8; ++r)
            Ps[(g * 8 + r) * 16 + loc] = (_Float16)pv[r];
        __syncthreads();
        v16h pa, bv0, bv1, bv2;
        #pragma unroll
        for (int i = 0; i < 16; ++i) {
            int v = i >> 1, p = i & 1;
            int ka = g * 16 + v * 2 + p;
            pa[i] = (ka < 16) ? Ps[loc * 16 + ka] : (_Float16)0.f;
            int kb = g * 16 + v * 2 + p;             // key index, 16..31 pad
            bv0[i] = (kb < 16) ? Vs[buf][kb * 48 + 0  + loc] : (_Float16)0.f;
            bv1[i] = (kb < 16) ? Vs[buf][kb * 48 + 16 + loc] : (_Float16)0.f;
            bv2[i] = (kb < 16) ? Vs[buf][kb * 48 + 32 + loc] : (_Float16)0.f;
        }
        o0 = WMMA_F16(pa, bv0, o0);
        o1 = WMMA_F16(pa, bv1, o1);
        o2 = WMMA_F16(pa, bv2, o2);
    }
    #pragma unroll
    for (int r = 0; r < 8; ++r) {
        int s = qt * 16 + g * 8 + r;
        float inv = 1.f / lrow[r];
        size_t base = (size_t)(s * L + n) * 96 + h * 48;
        oh[base + 0  + loc] = (_Float16)(o0[r] * inv);
        oh[base + 16 + loc] = (_Float16)(o1[r] * inv);
        oh[base + 32 + loc] = (_Float16)(o2[r] * inv);
    }
}

// ---------------------------------------------------------------------------
__global__ void f32_to_f16_kernel(const float* __restrict__ src,
                                  _Float16* __restrict__ dst, int n) {
    int i = blockIdx.x * blockDim.x + threadIdx.x;
    if (i < n) dst[i] = (_Float16)src[i];
}

// conv9 + embedding concat + positional encoding -> x (f32) and f16 copy
__global__ void embed_kernel(const float* __restrict__ in_series,
                             const int* __restrict__ ctx,
                             const float* __restrict__ conv_w,
                             const float* __restrict__ conv_b,
                             const float* __restrict__ emb0,
                             const float* __restrict__ emb1,
                             float* __restrict__ x, _Float16* __restrict__ xh,
                             int B, int L)
{
    int idx = blockIdx.x * blockDim.x + threadIdx.x;
    if (idx >= B * L) return;
    int b = idx / L, l = idx % L;
    size_t base = (size_t)idx * 96;
    float inbuf[9];
    #pragma unroll
    for (int t = 0; t < 9; ++t) {
        int p = l + t - 4;
        inbuf[t] = (p >= 0 && p < L) ? in_series[(size_t)b * L + p] : 0.f;
    }
    int c0 = ctx[b * 2 + 0], c1 = ctx[b * 2 + 1];
    const float nl = -9.210340371976184f / 96.f;   // -ln(10000)/E
    for (int e = 0; e < 96; ++e) {
        float val;
        if (e < 64) {
            float acc = conv_b[e];
            #pragma unroll
            for (int t = 0; t < 9; ++t) acc += conv_w[e * 9 + t] * inbuf[t];
            val = acc;
        } else if (e < 80) {
            val = emb0[c0 * 16 + (e - 64)];
        } else {
            val = emb1[c1 * 16 + (e - 80)];
        }
        int j = e >> 1;
        float div = __expf((float)(2 * j) * nl);
        float pe = (e & 1) ? __cosf((float)l * div) : __sinf((float)l * div);
        val += pe;
        x[base + e] = val;
        xh[base + e] = (_Float16)val;
    }
}

// Fused residual + LayerNorm over E=96; one 128-thread block per row
__global__ __launch_bounds__(128) void ln_kernel(
    const float* __restrict__ xin, const float* __restrict__ add,
    const float* __restrict__ w, const float* __restrict__ b,
    float* __restrict__ xout, _Float16* __restrict__ xh, int E)
{
    __shared__ float red[128];
    int r = blockIdx.x, t = threadIdx.x;
    float v = 0.f;
    if (t < E) v = xin[(size_t)r * E + t] + add[(size_t)r * E + t];
    red[t] = (t < E) ? v : 0.f;
    __syncthreads();
    for (int s = 64; s > 0; s >>= 1) { if (t < s) red[t] += red[t + s]; __syncthreads(); }
    float mean = red[0] / (float)E;
    __syncthreads();
    float dv = (t < E) ? (v - mean) : 0.f;
    red[t] = dv * dv;
    __syncthreads();
    for (int s = 64; s > 0; s >>= 1) { if (t < s) red[t] += red[t + s]; __syncthreads(); }
    float rstd = rsqrtf(red[0] / (float)E + 1e-5f);
    if (t < E) {
        float y = dv * rstd * w[t] + b[t];
        xout[(size_t)r * E + t] = y;
        xh[(size_t)r * E + t] = (_Float16)y;
    }
}

// KNN feature block: one block per (batch element, side)
#define KNN_T 128
__global__ __launch_bounds__(KNN_T) void knn_kernel(
    const float* __restrict__ emb, const int* __restrict__ ctx,
    const float* __restrict__ residuals, float* __restrict__ feats,
    int side, int B, int L, int N, int EMB)
{
    __shared__ float wv[1000];
    __shared__ float ecen[16];
    __shared__ float rmax[KNN_T];
    __shared__ int   rarg[KNN_T];
    __shared__ int   selI[21];
    __shared__ float selW[21];
    int b = blockIdx.x, t = threadIdx.x;
    int idx   = ctx[b * 2 + side];
    int other = ctx[b * 2 + (1 - side)];
    if (t < EMB) ecen[t] = emb[idx * EMB + t];
    __syncthreads();
    for (int n = t; n < N; n += KNN_T) {
        float d2 = 0.f;
        for (int e = 0; e < EMB; ++e) { float df = ecen[e] - emb[n * EMB + e]; d2 += df * df; }
        float d = sqrtf(fmaxf(d2, 0.f)) + 0.001f;
        if (n == idx) d = 0.f;
        wv[n] = __expf(-d);           // TAU = 1
    }
    __syncthreads();
    for (int it = 0; it < 21; ++it) {
        float best = -2.f; int barg = N;
        for (int n = t; n < N; n += KNN_T) {
            float v = wv[n];
            if (v > best) { best = v; barg = n; }
        }
        rmax[t] = best; rarg[t] = barg;
        __syncthreads();
        for (int s = KNN_T / 2; s > 0; s >>= 1) {
            if (t < s) {
                bool take = rmax[t + s] > rmax[t] ||
                            (rmax[t + s] == rmax[t] && rarg[t + s] < rarg[t]);
                if (take) { rmax[t] = rmax[t + s]; rarg[t] = rarg[t + s]; }
            }
            __syncthreads();
        }
        if (t == 0) { selI[it] = rarg[0]; selW[it] = rmax[0]; wv[rarg[0]] = -1.f; }
        __syncthreads();
    }
    for (int l = t; l < L; l += KNN_T) {
        float f1 = 0.f, fw = 0.f, ssum = 0.f;
        float sv[20];
        for (int k = 0; k < 20; ++k) {
            int nb = selI[k + 1];
            float s = (side == 0)
                ? residuals[((size_t)l * N + nb) * N + other]
                : residuals[((size_t)l * N + other) * N + nb];
            float wk  = selW[k + 1];
            float wkl = (s == 0.f) ? wk * 0.001f : wk;
            f1 += wkl; fw += wkl * s; ssum += s; sv[k] = s;
        }
        float f2 = fw / f1;
        float mean = ssum / 20.f;
        float var = 0.f;
        for (int k = 0; k < 20; ++k) { float d = sv[k] - mean; var += d * d; }
        float f3 = sqrtf(var / 19.f);      // ddof=1
        size_t fb = ((size_t)b * L + l) * 6;
        feats[fb + side]     = f1;
        feats[fb + 2 + side] = f2;
        feats[fb + 4 + side] = f3;
    }
}

// Head + masked MAE reduction. One thread per (b,l).
__global__ __launch_bounds__(256) void head_kernel(
    const float* __restrict__ x, const float* __restrict__ feats6,
    const float* __restrict__ out1_w, const float* __restrict__ out1_b,
    const float* __restrict__ mean_w, const float* __restrict__ mean_b,
    const float* __restrict__ out_series, const int* __restrict__ mask,
    const float* __restrict__ ctx_std, float* __restrict__ accum,
    int B, int L)
{
    __shared__ float red[256];
    int idx = blockIdx.x * blockDim.x + threadIdx.x;
    float dsum = 0.f, msum = 0.f;
    if (idx < B * L) {
        int b = idx / L;
        float f[102];
        #pragma unroll
        for (int j = 0; j < 6; ++j) f[j] = feats6[(size_t)idx * 6 + j];
        for (int j = 0; j < 96; ++j) f[6 + j] = fmaxf(x[(size_t)idx * 96 + j], 0.f);
        float mean = mean_b[0];
        for (int o = 0; o < 64; ++o) {
            float acc = out1_b[o];
            for (int j = 0; j < 102; ++j) acc += out1_w[o * 102 + j] * f[j];
            mean += fmaxf(acc, 0.f) * mean_w[o];
        }
        float m = (mask[idx] != 0) ? 1.f : 0.f;
        dsum = fabsf((out_series[idx] - mean) * ctx_std[b]) * m;
        msum = m;
    }
    int t = threadIdx.x;
    red[t] = dsum; __syncthreads();
    for (int s = 128; s > 0; s >>= 1) { if (t < s) red[t] += red[t + s]; __syncthreads(); }
    if (t == 0) atomicAdd(&accum[0], red[0]);
    __syncthreads();
    red[t] = msum; __syncthreads();
    for (int s = 128; s > 0; s >>= 1) { if (t < s) red[t] += red[t + s]; __syncthreads(); }
    if (t == 0) atomicAdd(&accum[1], red[0]);
}

__global__ void init_kernel(float* accum) {
    if (threadIdx.x < 2 && blockIdx.x == 0) accum[threadIdx.x] = 0.f;
}

__global__ void finalize_kernel(const float* accum, float* out) {
    if (threadIdx.x == 0 && blockIdx.x == 0) out[0] = accum[0] / accum[1];
}

// ---------------------------------------------------------------------------
extern "C" void kernel_launch(void* const* d_in, const int* in_sizes, int n_in,
                              void* d_out, int out_size, void* d_ws, size_t ws_size,
                              hipStream_t stream) {
    (void)in_sizes; (void)n_in; (void)out_size; (void)ws_size;
    const int B = 512, L = 128, E = 96, N = 1000, EMB = 16, NHID = 32, E3 = 288;
    const int NLAYERS = 4;
    const int M = B * L;

    const float* in_series  = (const float*)d_in[0];
    const float* out_series = (const float*)d_in[1];
    const int*   mask       = (const int*)  d_in[2];
    const int*   ctx        = (const int*)  d_in[3];
    const float* ctx_std    = (const float*)d_in[4];
    const float* residuals  = (const float*)d_in[5];
    const float* conv_w     = (const float*)d_in[6];
    const float* conv_b     = (const float*)d_in[7];
    const float* emb0       = (const float*)d_in[8];
    const float* emb1       = (const float*)d_in[9];
    const float* in_proj_w  = (const float*)d_in[10];
    const float* in_proj_b  = (const float*)d_in[11];
    const float* out_proj_w = (const float*)d_in[12];
    const float* out_proj_b = (const float*)d_in[13];
    const float* ln1_w      = (const float*)d_in[14];
    const float* ln1_b      = (const float*)d_in[15];
    const float* ff1_w      = (const float*)d_in[16];
    const float* ff1_b      = (const float*)d_in[17];
    const float* ff2_w      = (const float*)d_in[18];
    const float* ff2_b      = (const float*)d_in[19];
    const float* ln2_w      = (const float*)d_in[20];
    const float* ln2_b      = (const float*)d_in[21];
    const float* out1_w     = (const float*)d_in[22];
    const float* out1_b     = (const float*)d_in[23];
    const float* mean_w     = (const float*)d_in[24];
    const float* mean_b     = (const float*)d_in[25];

    // workspace carving (256B aligned blocks)
    char* ws = (char*)d_ws;
    size_t off = 0;
    auto carve = [&](size_t bytes) {
        void* p = ws + off;
        off += (bytes + 255) & ~(size_t)255;
        return p;
    };
    float*     accum  = (float*)    carve(2 * sizeof(float));
    float*     x      = (float*)    carve((size_t)M * E * 4);
    _Float16*  xh     = (_Float16*) carve((size_t)M * E * 2);
    _Float16*  qkvh   = (_Float16*) carve((size_t)M * E3 * 2);
    _Float16*  oh     = (_Float16*) carve((size_t)M * E * 2);
    float*     tmp    = (float*)    carve((size_t)M * E * 4);
    _Float16*  h1h    = (_Float16*) carve((size_t)M * NHID * 2);
    float*     feats6 = (float*)    carve((size_t)M * 6 * 4);
    // f16 weight copies (enables pure async B-tile staging)
    const int nwi = NLAYERS * E3 * E, nwo = NLAYERS * E * E;
    const int nw1 = NLAYERS * NHID * E, nw2 = NLAYERS * E * NHID;
    _Float16*  whi = (_Float16*) carve((size_t)nwi * 2);
    _Float16*  who = (_Float16*) carve((size_t)nwo * 2);
    _Float16*  wf1 = (_Float16*) carve((size_t)nw1 * 2);
    _Float16*  wf2 = (_Float16*) carve((size_t)nw2 * 2);

    init_kernel<<<1, 32, 0, stream>>>(accum);
    f32_to_f16_kernel<<<(nwi + 255) / 256, 256, 0, stream>>>(in_proj_w,  whi, nwi);
    f32_to_f16_kernel<<<(nwo + 255) / 256, 256, 0, stream>>>(out_proj_w, who, nwo);
    f32_to_f16_kernel<<<(nw1 + 255) / 256, 256, 0, stream>>>(ff1_w,      wf1, nw1);
    f32_to_f16_kernel<<<(nw2 + 255) / 256, 256, 0, stream>>>(ff2_w,      wf2, nw2);
    embed_kernel<<<(M + 255) / 256, 256, 0, stream>>>(
        in_series, ctx, conv_w, conv_b, emb0, emb1, x, xh, B, L);

    for (int l = 0; l < NLAYERS; ++l) {
        // QKV projection (f16 out for attention)
        gemm_wmma<<<dim3(E3 / 16, M / 16), 32, 0, stream>>>(
            xh, whi + (size_t)l * E3 * E, in_proj_b + (size_t)l * E3,
            nullptr, qkvh, M, E3, E, 0);
        // attention across the batch axis (S=B tokens per position n)
        attn_wmma<<<dim3(B / 16, L, 2), 32, 0, stream>>>(qkvh, oh, B, L, E3);
        // output projection
        gemm_wmma<<<dim3(E / 16, M / 16), 32, 0, stream>>>(
            oh, who + (size_t)l * E * E, out_proj_b + (size_t)l * E,
            tmp, nullptr, M, E, E, 0);
        // residual + LN1
        ln_kernel<<<M, 128, 0, stream>>>(x, tmp, ln1_w + (size_t)l * E,
                                         ln1_b + (size_t)l * E, x, xh, E);
        // FF1 with ReLU (f16 out)
        gemm_wmma<<<dim3(NHID / 16, M / 16), 32, 0, stream>>>(
            xh, wf1 + (size_t)l * NHID * E, ff1_b + (size_t)l * NHID,
            nullptr, h1h, M, NHID, E, 1);
        // FF2
        gemm_wmma<<<dim3(E / 16, M / 16), 32, 0, stream>>>(
            h1h, wf2 + (size_t)l * E * NHID, ff2_b + (size_t)l * E,
            tmp, nullptr, M, E, NHID, 0);
        // residual + LN2
        ln_kernel<<<M, 128, 0, stream>>>(x, tmp, ln2_w + (size_t)l * E,
                                         ln2_b + (size_t)l * E, x, xh, E);
    }

    knn_kernel<<<B, KNN_T, 0, stream>>>(emb0, ctx, residuals, feats6, 0, B, L, N, EMB);
    knn_kernel<<<B, KNN_T, 0, stream>>>(emb1, ctx, residuals, feats6, 1, B, L, N, EMB);

    head_kernel<<<(M + 255) / 256, 256, 0, stream>>>(
        x, feats6, out1_w, out1_b, mean_w, mean_b,
        out_series, mask, ctx_std, accum, B, L);
    finalize_kernel<<<1, 32, 0, stream>>>(accum, (float*)d_out);
}
